// MambaVisionMixer_41360535060766
// MI455X (gfx1250) — matrix operand order repaired and checked by reading, not compiled, not run on hardware
//
#include <hip/hip_runtime.h>
#include <hip/hip_bf16.h>
#include <stdint.h>

typedef __attribute__((ext_vector_type(16))) __bf16 v16bf;
typedef __attribute__((ext_vector_type(8)))  float  v8f;

union Frag32 { float4 f[2]; v16bf v; };

__device__ __forceinline__ uint16_t f2bf(float x) {
  uint32_t u = __float_as_uint(x);
  u += 0x7FFFu + ((u >> 16) & 1u);          // round-to-nearest-even
  return (uint16_t)(u >> 16);
}
__device__ __forceinline__ float bf2f(uint16_t h) {
  return __uint_as_float(((uint32_t)h) << 16);
}

// CDNA5 async global->LDS (ASYNCcnt-tracked, bypasses VGPRs).
// VDST = LDS byte address VGPR, VADDR = 64-bit global address (GV mode).
__device__ __forceinline__ void async_b128(uint32_t lds, const void* g) {
  asm volatile("global_load_async_to_lds_b128 %0, %1, off"
               :: "v"(lds), "v"(g) : "memory");
}
__device__ __forceinline__ void wait_async0() {
  asm volatile("s_wait_asynccnt 0" ::: "memory");
}

// ------------------------------------------------------------------
// bf16 WMMA GEMM:  C[M,N] = A[M,K] * B[N,K]^T  (f32 accum)
// 128 threads (4 waves); block tile 128x64; K-step 32.
// Wave w: rows [32w,32w+32) -> 2 A-frags x 4 B-frags = 8 WMMA / K-step.
// Double-buffered LDS fed by async loads (overlap load of tile k+1
// with WMMA on tile k).
// ------------------------------------------------------------------
__global__ void __launch_bounds__(128)
gemm_bf16_wmma(const uint16_t* __restrict__ A, const uint16_t* __restrict__ B,
               float* __restrict__ C, int M, int N, int K) {
  constexpr int LDT = 40;                    // padded LDS row stride (elements)
  __shared__ __align__(16) uint16_t As[2][128 * LDT];
  __shared__ __align__(16) uint16_t Bs[2][64 * LDT];
  const int tid  = threadIdx.x;
  const int lane = tid & 31;
  const int wave = tid >> 5;
  const int m0 = blockIdx.y * 128;
  const int n0 = blockIdx.x * 64;
  const int mlow  = lane & 15;
  const int mhalf = lane >> 4;               // 0/1

  v8f acc[2][4] = {};

  auto issue = [&](int k0, int buf) {
    // A tile: 128 rows x 32 bf16 = 512 x 16B chunks (4 per thread)
#pragma unroll
    for (int p = 0; p < 4; ++p) {
      const int ch = tid + p * 128;
      const int r  = ch >> 2;
      const int o  = (ch & 3) * 8;
      async_b128((uint32_t)(uintptr_t)&As[buf][r * LDT + o],
                 &A[(size_t)(m0 + r) * K + k0 + o]);
    }
    // B tile: 64 rows x 32 bf16 = 256 x 16B chunks (2 per thread)
#pragma unroll
    for (int p = 0; p < 2; ++p) {
      const int ch = tid + p * 128;
      const int r  = ch >> 2;
      const int o  = (ch & 3) * 8;
      async_b128((uint32_t)(uintptr_t)&Bs[buf][r * LDT + o],
                 &B[(size_t)(n0 + r) * K + k0 + o]);
    }
  };

  auto compute = [&](int buf) {
    // A frags 16x32: lanes 0-15 rows, kb = 0/8 by lane half
    Frag32 af0, af1;
    {
      const uint16_t* base0 = &As[buf][(wave * 32 + mlow) * LDT];
      const uint16_t* base1 = base0 + 16 * LDT;
      const int kb = mhalf * 8;
      af0.f[0] = *(const float4*)(base0 + kb);
      af0.f[1] = *(const float4*)(base0 + 16 + kb);
      af1.f[0] = *(const float4*)(base1 + kb);
      af1.f[1] = *(const float4*)(base1 + 16 + kb);
    }
#pragma unroll
    for (int j = 0; j < 4; ++j) {
      Frag32 bfr;                            // B frag 32x16: kb = 0/8 by lane half
      const uint16_t* base = &Bs[buf][(j * 16 + mlow) * LDT + mhalf * 8];
      bfr.f[0] = *(const float4*)(base);
      bfr.f[1] = *(const float4*)(base + 16);
      acc[0][j] = __builtin_amdgcn_wmma_f32_16x16x32_bf16(
          false, af0.v, false, bfr.v, (short)0, acc[0][j], false, false);
      acc[1][j] = __builtin_amdgcn_wmma_f32_16x16x32_bf16(
          false, af1.v, false, bfr.v, (short)0, acc[1][j], false, false);
    }
  };

  issue(0, 0);
  wait_async0();
  __syncthreads();
  int buf = 0;
  int k0 = 0;
  for (; k0 + 32 < K; k0 += 32) {
    issue(k0 + 32, buf ^ 1);                 // prefetch next tile (async)
    compute(buf);                            // WMMA on current tile
    wait_async0();
    __syncthreads();
    buf ^= 1;
  }
  compute(buf);                              // tail tile

  // C/D layout: lane 0-15 -> N=lane, M=reg; lane 16-31 -> M=8+reg
  const int ncol = n0 + mlow;
#pragma unroll
  for (int mi = 0; mi < 2; ++mi) {
    const int mrow = m0 + wave * 32 + mi * 16 + mhalf * 8;
#pragma unroll
    for (int j = 0; j < 4; ++j)
#pragma unroll
      for (int r = 0; r < 8; ++r)
        C[(size_t)(mrow + r) * N + ncol + j * 16] = acc[mi][j][r];
  }
}

// ------------------------------------------------------------------
// elementwise conversions / padding
// ------------------------------------------------------------------
__global__ void __launch_bounds__(256)
k_f32_to_bf16(const float* __restrict__ s, uint16_t* __restrict__ d, int n) {
  int i = blockIdx.x * 256 + threadIdx.x;
  if (i < n) d[i] = f2bf(s[i]);
}

__global__ void __launch_bounds__(256)
k_pad_wx(const float* __restrict__ s, uint16_t* __restrict__ d) {   // [80,768]->[128,768]
  int i = blockIdx.x * 256 + threadIdx.x;
  if (i >= 128 * 768) return;
  int r = i / 768, c = i % 768;
  d[i] = (r < 80) ? f2bf(s[r * 768 + c]) : (uint16_t)0;
}

__global__ void __launch_bounds__(256)
k_pad_wdt(const float* __restrict__ s, uint16_t* __restrict__ d) {  // [768,48]->[768,64]
  int i = blockIdx.x * 256 + threadIdx.x;
  if (i >= 768 * 64) return;
  int r = i / 64, c = i % 64;
  d[i] = (c < 48) ? f2bf(s[r * 48 + c]) : (uint16_t)0;
}

__global__ void __launch_bounds__(256)
k_make_dt(const float* __restrict__ xdbl, uint16_t* __restrict__ d) { // dt cols of x_dbl
  int i = blockIdx.x * 256 + threadIdx.x;
  if (i >= 8192 * 64) return;
  int r = i / 64, c = i % 64;
  d[i] = (c < 48) ? f2bf(xdbl[(size_t)r * 128 + c]) : (uint16_t)0;
}

// ------------------------------------------------------------------
// direction maps: sequence position s -> row-major token index
// dir0 row-fwd, dir1 row-bwd, dir2 col-fwd, dir3 col-bwd  (32x32 grid)
// ------------------------------------------------------------------
__device__ __forceinline__ int seq2tok(int dir, int s) {
  switch (dir) {
    case 0: return s;
    case 1: return 1023 - s;
    case 2: return ((s & 31) << 5) | (s >> 5);
    default: { int ss = 1023 - s; return ((ss & 31) << 5) | (ss >> 5); }
  }
}

// depthwise conv K=4 ('same': pad left 1, right 2) + SiLU, per direction.
__global__ void __launch_bounds__(256)
k_conv_silu(const float* __restrict__ XZ, const float* __restrict__ wx,
            const float* __restrict__ wz, uint16_t* __restrict__ XCbf,
            float* __restrict__ ZS) {
  int i = blockIdx.x * 256 + threadIdx.x;        // over 4*2*1024*1536
  if (i >= 4 * 2 * 1024 * 1536) return;
  int c = i % 1536;
  int r = i / 1536;                              // (dir*2+b)*1024 + t
  int t = r % 1024;
  int db = r / 1024;
  int b = db & 1, dir = db >> 1;
  const float* w = (c < 768) ? (wx + c * 4) : (wz + (c - 768) * 4);
  float a = 0.f;
#pragma unroll
  for (int j = 0; j < 4; ++j) {
    int s = t - 1 + j;
    if (0 <= s && s < 1024) {
      int tok = seq2tok(dir, s);
      a += w[j] * XZ[(size_t)(b * 1024 + tok) * 1536 + c];
    }
  }
  float v = a / (1.f + __expf(-a));              // SiLU
  size_t ro = (size_t)r * 768;
  if (c < 768) XCbf[ro + c] = f2bf(v);
  else         ZS[ro + (c - 768)] = v;
}

// ------------------------------------------------------------------
// selective scan: one lane per (dir,b,channel); 16 states in registers.
// ------------------------------------------------------------------
__global__ void __launch_bounds__(256)
k_scan(const float* __restrict__ DELTA, const uint16_t* __restrict__ XCbf,
       const float* __restrict__ XDBL, const float* __restrict__ A_log,
       const float* __restrict__ Dvec, const float* __restrict__ bdt,
       float* __restrict__ Y) {
  int dir = blockIdx.x / 6;
  int ch  = (blockIdx.x % 6) * 256 + threadIdx.x;  // 0..1535 (b uniform per block)
  int b = ch / 768, d = ch % 768;
  float a[16], h[16];
#pragma unroll
  for (int n = 0; n < 16; ++n) { a[n] = -__expf(A_log[d * 16 + n]); h[n] = 0.f; }
  const float Dp = Dvec[d];
  const float bb = bdt[d];
  const size_t rowbase = (size_t)(dir * 2 + b) * 1024;
  for (int t = 0; t < 1024; ++t) {
    const size_t row = rowbase + t;
    if (t + 8 < 1024) {                            // stream-ahead prefetch
      __builtin_prefetch(&DELTA[(row + 8) * 768 + d], 0, 3);
      __builtin_prefetch(&XDBL[(row + 8) * 128 + 48], 0, 3);
      __builtin_prefetch(&XCbf[(row + 8) * 768 + d], 0, 3);
    }
    float x = DELTA[row * 768 + d] + bb;
    float delta = (x > 20.f) ? x : __logf(1.f + __expf(x));   // softplus
    float u = bf2f(XCbf[row * 768 + d]);
    float Bv[16], Cv[16];
    const float4* p = (const float4*)(XDBL + row * 128 + 48); // B: 48..63, C: 64..79
#pragma unroll
    for (int q = 0; q < 4; ++q) {
      *(float4*)(&Bv[q * 4]) = p[q];
      *(float4*)(&Cv[q * 4]) = p[q + 4];
    }
    const float dbu = delta * u;
    float accv = 0.f;
#pragma unroll
    for (int n = 0; n < 16; ++n) {
      float e = __expf(delta * a[n]);              // v_exp_f32 (TRANS, co-exec)
      h[n] = e * h[n] + dbu * Bv[n];
      accv += h[n] * Cv[n];
    }
    Y[row * 768 + d] = accv + u * Dp;
  }
}

// sum 4 directions back into row-major token order, cat [y|z], cast bf16
__global__ void __launch_bounds__(256)
k_accum(const float* __restrict__ Y, const float* __restrict__ ZS,
        uint16_t* __restrict__ Ysum) {
  int i = blockIdx.x * 256 + threadIdx.x;        // over 2048*1536
  if (i >= 2048 * 1536) return;
  int c = i % 1536;
  int row = i / 1536;                            // b*1024 + tok
  int b = row >> 10, tok = row & 1023;
  int tt = ((tok & 31) << 5) | (tok >> 5);       // transpose (self-inverse)
  int tdir[4] = { tok, 1023 - tok, tt, 1023 - tt };
  const float* src = (c < 768) ? Y : ZS;
  int cc = (c < 768) ? c : c - 768;
  float s = 0.f;
#pragma unroll
  for (int dir = 0; dir < 4; ++dir) {
    size_t rr = (size_t)((dir * 2 + b) * 1024 + tdir[dir]);
    s += src[rr * 768 + cc];
  }
  Ysum[(size_t)row * 1536 + c] = f2bf(s);
}

// ------------------------------------------------------------------
extern "C" void kernel_launch(void* const* d_in, const int* in_sizes, int n_in,
                              void* d_out, int out_size, void* d_ws, size_t ws_size,
                              hipStream_t stream) {
  const float* H    = (const float*)d_in[0];   // (2,1024,768)
  const float* Win  = (const float*)d_in[1];   // (1536,768)
  const float* wcx  = (const float*)d_in[2];   // (768,1,4)
  const float* wcz  = (const float*)d_in[3];   // (768,1,4)
  const float* Wx   = (const float*)d_in[4];   // (80,768)
  const float* Wdt  = (const float*)d_in[5];   // (768,48)
  const float* bdt  = (const float*)d_in[6];   // (768,)
  const float* Alog = (const float*)d_in[7];   // (768,16)
  const float* Dv   = (const float*)d_in[8];   // (768,)
  const float* Wout = (const float*)d_in[9];   // (768,1536)
  float* out = (float*)d_out;                  // (2,1024,768) f32
  char* ws = (char*)d_ws;

  size_t off = 0;
  auto take = [&](size_t bytes) {
    size_t o = off; off = (off + bytes + 255) & ~(size_t)255; return o;
  };
  uint16_t* Hbf    = (uint16_t*)(ws + take(2048ull * 768 * 2));
  uint16_t* Winbf  = (uint16_t*)(ws + take(1536ull * 768 * 2));
  uint16_t* Wxbf   = (uint16_t*)(ws + take(128ull * 768 * 2));
  uint16_t* Wdtbf  = (uint16_t*)(ws + take(768ull * 64 * 2));
  uint16_t* Woutbf = (uint16_t*)(ws + take(768ull * 1536 * 2));
  float*    XZ     = (float*)   (ws + take(2048ull * 1536 * 4));
  uint16_t* XCbf   = (uint16_t*)(ws + take(8192ull * 768 * 2));
  float*    ZS     = (float*)   (ws + take(8192ull * 768 * 4));
  float*    XDBL   = (float*)   (ws + take(8192ull * 128 * 4));
  uint16_t* DTbf   = (uint16_t*)(ws + take(8192ull * 64 * 2));
  float*    DELTA  = (float*)   (ws + take(8192ull * 768 * 4));
  float*    Y      = (float*)   (ws + take(8192ull * 768 * 4));
  uint16_t* Ysumbf = (uint16_t*)(ws + take(2048ull * 1536 * 2));

  // 1) fp32 -> bf16 conversion (with K/N padding to WMMA-friendly shapes)
  k_f32_to_bf16<<<6144, 256, 0, stream>>>(H, Hbf, 2048 * 768);
  k_f32_to_bf16<<<4608, 256, 0, stream>>>(Win, Winbf, 1536 * 768);
  k_pad_wx<<<384, 256, 0, stream>>>(Wx, Wxbf);
  k_pad_wdt<<<192, 256, 0, stream>>>(Wdt, Wdtbf);
  k_f32_to_bf16<<<4608, 256, 0, stream>>>(Wout, Woutbf, 768 * 1536);

  // 2) in-projection (shared by all 4 directions): XZ[2048,1536]
  gemm_bf16_wmma<<<dim3(1536 / 64, 2048 / 128), 128, 0, stream>>>(
      Hbf, Winbf, XZ, 2048, 1536, 768);

  // 3) per-direction depthwise conv + SiLU
  k_conv_silu<<<49152, 256, 0, stream>>>(XZ, wcx, wcz, XCbf, ZS);

  // 4) x_dbl for all 4 dirs batched: [8192,128] (cols 0..47 dt, 48..63 B, 64..79 C)
  gemm_bf16_wmma<<<dim3(128 / 64, 8192 / 128), 128, 0, stream>>>(
      XCbf, Wxbf, XDBL, 8192, 128, 768);

  // 5) delta projection: DELTA[8192,768] = dt @ W_dt^T (softplus fused in scan)
  k_make_dt<<<2048, 256, 0, stream>>>(XDBL, DTbf);
  gemm_bf16_wmma<<<dim3(768 / 64, 8192 / 128), 128, 0, stream>>>(
      DTbf, Wdtbf, DELTA, 8192, 768, 64);

  // 6) selective scan, 4 dirs x 1536 channels (192 wave32s)
  k_scan<<<24, 256, 0, stream>>>(DELTA, XCbf, XDBL, Alog, Dv, bdt, Y);

  // 7) fold directions back to token order, cat [y|z]
  k_accum<<<12288, 256, 0, stream>>>(Y, ZS, Ysumbf);

  // 8) out-projection straight into d_out
  gemm_bf16_wmma<<<dim3(768 / 64, 2048 / 128), 128, 0, stream>>>(
      Ysumbf, Woutbf, out, 2048, 768, 1536);
}
